// XASNet_NNconv_12996571037719
// MI455X (gfx1250) — matrix operand order repaired and checked by reading, not compiled, not run on hardware
//
#include <hip/hip_runtime.h>
#include <hip/hip_bf16.h>

typedef _Float16 h16;
typedef __attribute__((ext_vector_type(16))) _Float16 v16h;
typedef __attribute__((ext_vector_type(8)))  _Float16 v8h;
typedef __attribute__((ext_vector_type(8)))  float    v8f;

#define NNODES 4096
#define NEDGES 16384
#define NGRAPH 128
#define TOUT   100

// ---------------------------------------------------------------- utilities
__global__ void zero_f32_kernel(float* __restrict__ p, int n) {
    int i = blockIdx.x * blockDim.x + threadIdx.x;
    if (i < n) p[i] = 0.f;
}

// z[e,j] = relu(edge_attr[e,:] @ W1[:,j] + b1[j]),  j in [0,32)
__global__ void edge_z_kernel(const float* __restrict__ ea,
                              const float* __restrict__ W1,
                              const float* __restrict__ b1,
                              float* __restrict__ z) {
    int id = blockIdx.x * blockDim.x + threadIdx.x;
    if (id >= NEDGES * 32) return;
    int e = id >> 5, j = id & 31;
    float v = ea[e*3+0]*W1[0*32+j] + ea[e*3+1]*W1[1*32+j] + ea[e*3+2]*W1[2*32+j] + b1[j];
    z[id] = v > 0.f ? v : 0.f;
}

// Transposed f16 weight: Wt[o][K], K = k*CIN + i for k<32 (W2 term),
// K in [32*CIN, 33*CIN) carries b2, rest zero-padded to KPAD.
template<int CIN, int COUT, int KPAD>
__global__ void prep_wt_kernel(const float* __restrict__ W2,
                               const float* __restrict__ b2,
                               h16* __restrict__ Wt) {
    int id = blockIdx.x * blockDim.x + threadIdx.x;
    if (id >= COUT * KPAD) return;
    int o = id / KPAD, K = id - o * KPAD;
    float v = 0.f;
    if (K < 32 * CIN) {
        int k = K / CIN, i = K - k * CIN;
        v = W2[(size_t)(k * CIN + i) * COUT + o];
    } else if (K < 33 * CIN) {
        v = b2[(size_t)(K - 32 * CIN) * COUT + o];
    }
    Wt[(size_t)o * KPAD + K] = (h16)v;
}

__global__ void count_kernel(const int* __restrict__ dst, float* __restrict__ cnt) {
    int e = blockIdx.x * blockDim.x + threadIdx.x;
    if (e < NEDGES) atomicAdd(&cnt[dst[e]], 1.f);
}

// --------------------------------------------- generic NNConv (layer 1 only)
// LDS-staged A-tile path for CIN not divisible by 32 (tiny: Kpad=192).
template<int CIN, int COUT, int NW, int KPAD>
__global__ __launch_bounds__(128)
void nnconv_wmma_kernel(const int*   __restrict__ src,
                        const int*   __restrict__ dst,
                        const float* __restrict__ hin,
                        const float* __restrict__ z,
                        const h16*   __restrict__ Wt,
                        float*       __restrict__ ssum) {
    __shared__ h16   Atile[16 * 32];
    __shared__ float zs[16 * 32];
    __shared__ float hs[16 * CIN];
    __shared__ int   dl[16];

    const int tid  = threadIdx.x;
    const int lane = tid & 31;
    const int wave = tid >> 5;
    const int e0   = blockIdx.x * 16;

    if (tid < 16) dl[tid] = dst[e0 + tid];
    for (int i = tid; i < 16 * 32; i += 128)
        zs[i] = z[(size_t)(e0 + (i >> 5)) * 32 + (i & 31)];
    for (int i = tid; i < 16 * CIN; i += 128) {
        int m = i / CIN, c = i - m * CIN;
        hs[i] = hin[(size_t)src[e0 + m] * CIN + c];
    }
    __syncthreads();

    v8f zv = {};
    v8f acc[NW];
#pragma unroll
    for (int t = 0; t < NW; ++t) acc[t] = zv;

    const int m_a   = lane & 15;
    const int abase = (lane < 16) ? 0 : 8;
    const int kbase = (lane < 16) ? 0 : 16;
    const int ncol  = lane & 15;

    for (int kt = 0; kt < KPAD / 32; ++kt) {
        const int K0 = kt * 32;
#pragma unroll
        for (int c = 0; c < 4; ++c) {
            int idx = tid + c * 128;
            int m = idx & 15, kk = idx >> 4;
            int K = K0 + kk;
            float v = 0.f;
            if (K < 32 * CIN) {
                int k = K / CIN, i = K - k * CIN;
                v = zs[m * 32 + k] * hs[m * CIN + i];
            } else if (K < 33 * CIN) {
                v = hs[m * CIN + (K - 32 * CIN)];
            }
            Atile[m * 32 + kk] = (h16)v;
        }
        __syncthreads();

        v16h a;
        {
            v8h a0 = *(const v8h*)&Atile[m_a * 32 + abase];
            v8h a1 = *(const v8h*)&Atile[m_a * 32 + abase + 16];
#pragma unroll
            for (int j = 0; j < 8; ++j) { a[j] = a0[j]; a[8 + j] = a1[j]; }
        }
#pragma unroll
        for (int t = 0; t < NW; ++t) {
            const h16* bp = Wt + (size_t)(wave * NW * 16 + t * 16 + ncol) * KPAD + K0 + kbase;
            v16h b;
            {
                v8h b0 = *(const v8h*)(bp - kbase + abase);
                v8h b1 = *(const v8h*)(bp - kbase + abase + 16);
#pragma unroll
                for (int j = 0; j < 8; ++j) { b[j] = b0[j]; b[8 + j] = b1[j]; }
            }
            acc[t] = __builtin_amdgcn_wmma_f32_16x16x32_f16(
                false, a, false, b, (short)0, acc[t], false, false);
        }
        __syncthreads();
    }

#pragma unroll
    for (int t = 0; t < NW; ++t) {
#pragma unroll
        for (int r = 0; r < 8; ++r) {
            int m = r + ((lane & 16) ? 8 : 0);
            int n = wave * NW * 16 + t * 16 + ncol;
            atomicAdd(&ssum[(size_t)dl[m] * COUT + n], acc[t][r]);
        }
    }
}

// --------------------------------- fast NNConv (CIN % 32 == 0, layers 2 & 3)
// Exploits Kron structure: a 32-wide K block has constant k and contiguous i,
// so each lane's A fragment = scalar z[m,k0] * register-resident h-row chunk.
// No LDS traffic and no barriers in the K loop. Bias rows via z_ext[.,32]=1.
template<int CIN, int COUT, int NW>
__global__ __launch_bounds__(128)
void nnconv_wmma_fast_kernel(const int*   __restrict__ src,
                             const int*   __restrict__ dst,
                             const float* __restrict__ hin,
                             const float* __restrict__ z,
                             const h16*   __restrict__ Wt,
                             float*       __restrict__ ssum) {
    static_assert(CIN % 32 == 0, "fast path requires 32 | CIN");
    constexpr int KPAD = 33 * CIN;
    constexpr int NIB  = CIN / 32;

    __shared__ h16   hsh[16 * CIN];   // gathered h[src] rows, f16
    __shared__ float zsc[16 * 33];    // z rows + trailing 1.0 for bias block
    __shared__ int   dl[16];

    const int tid  = threadIdx.x;
    const int lane = tid & 31;
    const int wave = tid >> 5;
    const int e0   = blockIdx.x * 16;

    if (tid < 16) dl[tid] = dst[e0 + tid];
    for (int i = tid; i < 16 * 33; i += 128) {
        int m = i / 33, k = i - m * 33;
        zsc[i] = (k < 32) ? z[(size_t)(e0 + m) * 32 + k] : 1.0f;
    }
    for (int i = tid; i < 16 * CIN; i += 128) {
        int m = i / CIN, c = i - m * CIN;
        hsh[i] = (h16)hin[(size_t)src[e0 + m] * CIN + c];
    }
    __syncthreads();

    const int m_a   = lane & 15;
    const int abase = (lane < 16) ? 0 : 8;
    const int kbase = (lane < 16) ? 0 : 16;
    const int ncol  = lane & 15;

    // Hoist this lane's h-row chunks into registers (k0-invariant).
    const h16* hrow = &hsh[m_a * CIN];
    v8h ah0[NIB], ah1[NIB];
#pragma unroll
    for (int ib = 0; ib < NIB; ++ib) {
        ah0[ib] = *(const v8h*)(hrow + ib * 32 + abase);
        ah1[ib] = *(const v8h*)(hrow + ib * 32 + abase + 16);
    }

    v8f zv = {};
    v8f acc[NW];
#pragma unroll
    for (int t = 0; t < NW; ++t) acc[t] = zv;

    for (int k0 = 0; k0 < 33; ++k0) {
        const h16 zh = (h16)zsc[m_a * 33 + k0];
#pragma unroll
        for (int ib = 0; ib < NIB; ++ib) {
            const int K0 = k0 * CIN + ib * 32;
            v16h a;
#pragma unroll
            for (int j = 0; j < 8; ++j) {
                a[j]     = ah0[ib][j] * zh;   // packs to v_pk_mul_f16
                a[8 + j] = ah1[ib][j] * zh;
            }
#pragma unroll
            for (int t = 0; t < NW; ++t) {
                const h16* bp = Wt + (size_t)(wave * NW * 16 + t * 16 + ncol) * KPAD + K0 + kbase;
                v16h b;
                {
                    v8h b0 = *(const v8h*)(bp - kbase + abase);
                    v8h b1 = *(const v8h*)(bp - kbase + abase + 16);
#pragma unroll
                    for (int j = 0; j < 8; ++j) { b[j] = b0[j]; b[8 + j] = b1[j]; }
                }
                acc[t] = __builtin_amdgcn_wmma_f32_16x16x32_f16(
                    false, a, false, b, (short)0, acc[t], false, false);
            }
        }
    }

#pragma unroll
    for (int t = 0; t < NW; ++t) {
#pragma unroll
        for (int r = 0; r < 8; ++r) {
            int m = r + ((lane & 16) ? 8 : 0);
            int n = wave * NW * 16 + t * 16 + ncol;
            atomicAdd(&ssum[(size_t)dl[m] * COUT + n], acc[t][r]);
        }
    }
}

// ---------------------------------------------- per-node root/mean/BN/ReLU
template<int CIN, int COUT>
__global__ void node_finalize_kernel(const float* __restrict__ hin,
                                     const float* __restrict__ ssum,
                                     const float* __restrict__ cnt,
                                     const float* __restrict__ root,
                                     const float* __restrict__ bias,
                                     const float* __restrict__ g,
                                     const float* __restrict__ b,
                                     const float* __restrict__ mean,
                                     const float* __restrict__ var,
                                     float* __restrict__ hout) {
    int id = blockIdx.x * blockDim.x + threadIdx.x;
    if (id >= NNODES * COUT) return;
    int v = id / COUT, o = id - v * COUT;
    float agg = ssum[id] / fmaxf(cnt[v], 1.f);
    float dot = 0.f;
#pragma unroll 8
    for (int i = 0; i < CIN; ++i)
        dot = fmaf(hin[(size_t)v * CIN + i], root[(size_t)i * COUT + o], dot);
    float h  = dot + agg + bias[o];
    float bn = g[o] * (h - mean[o]) * rsqrtf(var[o] + 1e-5f) + b[o];
    hout[id] = bn > 0.f ? bn : 0.f;
}

// --------------------------------------------------------------- pool+head
__global__ void pool_kernel(const float* __restrict__ h,
                            const int*   __restrict__ bs,
                            float* __restrict__ psum,
                            float* __restrict__ pcnt) {
    int id = blockIdx.x * blockDim.x + threadIdx.x;
    if (id >= NNODES * 256) return;
    int v = id >> 8, c = id & 255;
    int gg = bs[v];
    atomicAdd(&psum[(size_t)gg * 256 + c], h[id]);
    if (c == 0) atomicAdd(&pcnt[gg], 1.f);
}

__global__ void head_kernel(const float* __restrict__ psum,
                            const float* __restrict__ pcnt,
                            const float* __restrict__ W,
                            const float* __restrict__ bb,
                            float* __restrict__ out) {
    int id = blockIdx.x * blockDim.x + threadIdx.x;
    if (id >= NGRAPH * TOUT) return;
    int gg = id / TOUT, t = id - gg * TOUT;
    float inv = 1.f / fmaxf(pcnt[gg], 1.f);
    float dot = 0.f;
#pragma unroll 8
    for (int c = 0; c < 256; ++c)
        dot = fmaf(psum[(size_t)gg * 256 + c] * inv, W[(size_t)c * TOUT + t], dot);
    float o = dot + bb[t];
    out[id] = o > 0.f ? o : 0.1f * o;
}

// ------------------------------------------------------------------ driver
extern "C" void kernel_launch(void* const* d_in, const int* in_sizes, int n_in,
                              void* d_out, int out_size, void* d_ws, size_t ws_size,
                              hipStream_t stream) {
    (void)in_sizes; (void)n_in; (void)out_size; (void)ws_size;

    const float* x   = (const float*)d_in[0];
    const int*   ei  = (const int*)d_in[1];
    const int*   src = ei;
    const int*   dst = ei + NEDGES;
    const float* ea  = (const float*)d_in[2];
    const int*   bs  = (const int*)d_in[3];
    const float* mlpW = (const float*)d_in[34];
    const float* mlpb = (const float*)d_in[35];

    char* ws = (char*)d_ws;
    size_t off = 0;
    auto carve = [&](size_t bytes) {
        void* p = ws + off;
        off += (bytes + 255) & ~(size_t)255;
        return p;
    };
    float* z    = (float*)carve((size_t)NEDGES * 32 * 4);
    h16*   Wt   = (h16*)  carve((size_t)256 * 4224 * 2);
    float* hA   = (float*)carve((size_t)NNODES * 256 * 4);
    float* hB   = (float*)carve((size_t)NNODES * 256 * 4);
    float* ssum = (float*)carve((size_t)NNODES * 256 * 4);
    float* cnt  = (float*)carve((size_t)NNODES * 4);
    float* psum = (float*)carve((size_t)NGRAPH * 256 * 4);
    float* pcnt = (float*)carve((size_t)NGRAPH * 4);

    auto zero = [&](float* p, int n) {
        zero_f32_kernel<<<(n + 255) / 256, 256, 0, stream>>>(p, n);
    };

    zero(cnt, NNODES);
    count_kernel<<<NEDGES / 256, 256, 0, stream>>>(dst, cnt);
    zero(psum, NGRAPH * 256);
    zero(pcnt, NGRAPH);

    // ---- layer 1: cin=5, cout=64, Kpad=192 (generic path)
    {
        const float *W1 = (const float*)d_in[4],  *b1 = (const float*)d_in[5];
        const float *W2 = (const float*)d_in[6],  *b2 = (const float*)d_in[7];
        const float *rt = (const float*)d_in[8],  *bi = (const float*)d_in[9];
        const float *bg = (const float*)d_in[10], *bb = (const float*)d_in[11];
        const float *bm = (const float*)d_in[12], *bv = (const float*)d_in[13];
        edge_z_kernel<<<(NEDGES * 32) / 256, 256, 0, stream>>>(ea, W1, b1, z);
        prep_wt_kernel<5, 64, 192><<<(64 * 192 + 255) / 256, 256, 0, stream>>>(W2, b2, Wt);
        zero(ssum, NNODES * 64);
        nnconv_wmma_kernel<5, 64, 1, 192><<<NEDGES / 16, 128, 0, stream>>>(src, dst, x, z, Wt, ssum);
        node_finalize_kernel<5, 64><<<(NNODES * 64) / 256, 256, 0, stream>>>(
            x, ssum, cnt, rt, bi, bg, bb, bm, bv, hA);
    }
    // ---- layer 2: cin=64, cout=128, Kpad=2112 (fast path)
    {
        const float *W1 = (const float*)d_in[14], *b1 = (const float*)d_in[15];
        const float *W2 = (const float*)d_in[16], *b2 = (const float*)d_in[17];
        const float *rt = (const float*)d_in[18], *bi = (const float*)d_in[19];
        const float *bg = (const float*)d_in[20], *bb = (const float*)d_in[21];
        const float *bm = (const float*)d_in[22], *bv = (const float*)d_in[23];
        edge_z_kernel<<<(NEDGES * 32) / 256, 256, 0, stream>>>(ea, W1, b1, z);
        prep_wt_kernel<64, 128, 2112><<<(128 * 2112 + 255) / 256, 256, 0, stream>>>(W2, b2, Wt);
        zero(ssum, NNODES * 128);
        nnconv_wmma_fast_kernel<64, 128, 2><<<NEDGES / 16, 128, 0, stream>>>(src, dst, hA, z, Wt, ssum);
        node_finalize_kernel<64, 128><<<(NNODES * 128) / 256, 256, 0, stream>>>(
            hA, ssum, cnt, rt, bi, bg, bb, bm, bv, hB);
    }
    // ---- layer 3: cin=128, cout=256, Kpad=4224 (fast path)
    {
        const float *W1 = (const float*)d_in[24], *b1 = (const float*)d_in[25];
        const float *W2 = (const float*)d_in[26], *b2 = (const float*)d_in[27];
        const float *rt = (const float*)d_in[28], *bi = (const float*)d_in[29];
        const float *bg = (const float*)d_in[30], *bb = (const float*)d_in[31];
        const float *bm = (const float*)d_in[32], *bv = (const float*)d_in[33];
        edge_z_kernel<<<(NEDGES * 32) / 256, 256, 0, stream>>>(ea, W1, b1, z);
        prep_wt_kernel<128, 256, 4224><<<(256 * 4224 + 255) / 256, 256, 0, stream>>>(W2, b2, Wt);
        zero(ssum, NNODES * 256);
        nnconv_wmma_fast_kernel<128, 256, 4><<<NEDGES / 16, 128, 0, stream>>>(src, dst, hB, z, Wt, ssum);
        node_finalize_kernel<128, 256><<<(NNODES * 256) / 256, 256, 0, stream>>>(
            hB, ssum, cnt, rt, bi, bg, bb, bm, bv, hA);
    }

    pool_kernel<<<(NNODES * 256) / 256, 256, 0, stream>>>(hA, bs, psum, pcnt);
    head_kernel<<<(NGRAPH * TOUT + 255) / 256, 256, 0, stream>>>(psum, pcnt, mlpW, mlpb, (float*)d_out);
}
